// GaussianOrthLayer_34986803593234
// MI455X (gfx1250) — matrix operand order repaired and checked by reading, not compiled, not run on hardware
//
#include <hip/hip_runtime.h>
#include <hip/hip_bf16.h>

// ---------------------------------------------------------------------------
// Problem constants (from the reference)
// ---------------------------------------------------------------------------
#define BB 2
#define NN 6
#define HG 28
#define WG 60
#define CC 128
#define LL (HG * WG)            // 1680
#define BN (BB * NN)            // 12
#define NLOC (BN * LL)          // 20160 (b*n*L locations)
#define P2 20                   // 5 * NUM_POINTS
#define NPTS (NLOC * P2)        // 403200 total sample points
#define NGROUPS (NPTS / 16)     // 25200 WMMA row-groups of 16 points
#define HID 256                 // 2*C hidden width
#define IMG_W 480.0f
#define IMG_H 224.0f

// d_out float offsets
#define OUT_FEATS 0
#define OUT_CAM   ((size_t)NPTS * CC)                 // 51,609,600
#define OUT_PTS   (OUT_CAM + (size_t)NPTS * 2)       // 52,416,000

// workspace byte offsets (all 128B aligned)
#define WS_FEATT  0                                   // 12*1680*128 f32 = 10,321,920 B
#define WS_PN4    10321920                            // 403200*4 f32   =  6,451,200 B
#define WS_PXPY   16773120                            // 403200*2 f32   =  3,225,600 B
#define WS_WB     19998720                            // 64*512 f16     =     65,536 B

typedef __attribute__((ext_vector_type(16))) _Float16 v16h;
typedef __attribute__((ext_vector_type(8)))  float    v8f;

__device__ __forceinline__ int iclamp(int v, int lo, int hi) {
    return v < lo ? lo : (v > hi ? hi : v);
}
__device__ __forceinline__ float fclamp01(float v) {
    return fminf(fmaxf(v, 0.f), 1.f);
}

// ---------------------------------------------------------------------------
// Kernel T: transpose img_features (BN,C,H,W) -> (BN,H,W,C) so bilinear
// gathers are coalesced over the channel dim (lanes = channels).
// ---------------------------------------------------------------------------
__global__ __launch_bounds__(256) void transpose_feat_kernel(
    const float* __restrict__ fin, float* __restrict__ fout)
{
    int idx = blockIdx.x * 256 + threadIdx.x;        // exact: 12*1680*128 / 256
    int c  = idx & (CC - 1);
    int t  = idx >> 7;                               // bn*1680 + hw
    int hw = t % LL;
    int bn = t / LL;
    fout[idx] = fin[((size_t)bn * CC + c) * LL + hw];
}

// ---------------------------------------------------------------------------
// Kernel W: pack W2 (256x128 f32, row-major) into fp16 WMMA B-fragments.
// v_wmma_f32_16x16x32_f16 B layout (32x16): half slot i<8 of a lane is
// K = khi*8 + i, slot i>=8 is K = 16 + khi*8 + (i-8).
// Flat layout: wB[(((kc*8+nt)*32 + lane)*16 + i)]
// ---------------------------------------------------------------------------
__global__ __launch_bounds__(256) void pack_w2_kernel(
    const float* __restrict__ W2, _Float16* __restrict__ wB)
{
    int t = blockIdx.x * 256 + threadIdx.x;          // exact: 32768 / 256
    int i    = t & 15;
    int lane = (t >> 4) & 31;
    int nt   = (t >> 9) & 7;
    int kc   = t >> 12;
    int n    = lane & 15;
    int khi  = lane >> 4;
    int k    = kc * 32 + (i < 8 ? khi * 8 + i : 16 + khi * 8 + (i - 8));
    int c    = nt * 16 + n;
    wB[t] = (_Float16)W2[(size_t)k * CC + c];
}

// ---------------------------------------------------------------------------
// Kernel A: geometry. One thread per (b,n,l) location -> 20 points.
// Writes pts + cam_xy straight to d_out; pn4 (clip-normalized + 1.0 pad) and
// pixel coords (px = cam_x/8 - 0.5, py = cam_y/8 - 0.5) to workspace.
// ---------------------------------------------------------------------------
__global__ __launch_bounds__(256) void geom_kernel(
    const float* __restrict__ means3D,
    const float* __restrict__ dvec,
    const float* __restrict__ cov3D,
    const float* __restrict__ l2i,
    float* __restrict__ cam_out,
    float* __restrict__ pts_out,
    float* __restrict__ pn4,
    float* __restrict__ pxpy)
{
    int loc = blockIdx.x * 256 + threadIdx.x;
    if (loc >= NLOC) return;
    int bn = loc / LL;

    const float* Mp = l2i + (size_t)bn * 16;
    float M[12];
    #pragma unroll
    for (int i = 0; i < 12; ++i) M[i] = Mp[i];       // rows 0..2 of lidar2img

    float mx = means3D[(size_t)loc * 3 + 0];
    float my = means3D[(size_t)loc * 3 + 1];
    float mz = means3D[(size_t)loc * 3 + 2];
    float dx = dvec[(size_t)loc * 3 + 0];
    float dy = dvec[(size_t)loc * 3 + 1];
    float dz = dvec[(size_t)loc * 3 + 2];
    const float* Cv = cov3D + (size_t)loc * 9;
    float q = dx * (Cv[0]*dx + Cv[1]*dy + Cv[2]*dz)
            + dy * (Cv[3]*dx + Cv[4]*dy + Cv[5]*dz)
            + dz * (Cv[6]*dx + Cv[7]*dy + Cv[8]*dz);
    float s = sqrtf(fmaxf(q, 0.f));
    float ox = dy, oy = -dx;                         // orth = (dv.y, -dv.x, 0)

    #pragma unroll
    for (int u = 0; u < 5; ++u) {
        float lu = -1.5f + 0.75f * (float)u;         // linspace(-1.5,1.5,5)
        float ux = fmaf(lu * s, dx, mx);
        float uy = fmaf(lu * s, dy, my);
        float uz = fmaf(lu * s, dz, mz);
        #pragma unroll
        for (int v = 0; v < 4; ++v) {
            float lv = -0.5f + (float)v * (1.0f / 3.0f);  // linspace(-.5,.5,4)
            float x = fmaf(lv, ox, ux);
            float y = fmaf(lv, oy, uy);
            float z = uz;
            float c0 = fmaf(M[0], x, fmaf(M[1], y, fmaf(M[2],  z, M[3])));
            float c1 = fmaf(M[4], x, fmaf(M[5], y, fmaf(M[6],  z, M[7])));
            float c2 = fmaf(M[8], x, fmaf(M[9], y, fmaf(M[10], z, M[11])));
            float h  = fmaxf(c2, 1e-6f);
            float cx = c0 / h, cy = c1 / h;

            size_t pi = (size_t)loc * P2 + (u * 4 + v);
            cam_out[pi * 2 + 0] = cx;
            cam_out[pi * 2 + 1] = cy;
            pts_out[pi * 3 + 0] = x;
            pts_out[pi * 3 + 1] = y;
            pts_out[pi * 3 + 2] = z;
            // pixel coords: (grid+1)*W/2-0.5 == cam/8 - 0.5 (480/60 = 224/28 = 8)
            pxpy[pi * 2 + 0] = cx * 0.125f - 0.5f;
            pxpy[pi * 2 + 1] = cy * 0.125f - 0.5f;
            pn4[pi * 4 + 0] = fclamp01((x + 50.f) * 0.01f);
            pn4[pi * 4 + 1] = fclamp01((y + 50.f) * 0.01f);
            pn4[pi * 4 + 2] = fclamp01((z + 4.f) * 0.125f);
            pn4[pi * 4 + 3] = 1.0f;                  // bias-padding lane
        }
    }
}

// ---------------------------------------------------------------------------
// Kernel B: the hot kernel. Each wave owns 16 points (one WMMA M-tile).
//   layer1 (K=4, VALU): hidden = relu([pn,1] . [W1;b1]) generated per lane
//     directly in the fp16 A-fragment layout (lane holds row M=lane&15; half
//     slots i<8 -> K = kc*32 + hi*8 + i, i>=8 -> K = kc*32 + 16 + hi*8 + i-8)
//   layer2: 8 K-chunks x 8 N-tiles of v_wmma_f32_16x16x32_f16 (64 WMMAs)
//   epilogue: D layout (lane = N, VGPR r = M = r + 8*hi) -> + b2 + bilinear
//     sample of the channel-major transposed feature map, coalesced store.
// No shared memory, no divergence before WMMA (EXEC all-ones guaranteed).
// ---------------------------------------------------------------------------
__global__ __launch_bounds__(256) void mlp_sample_kernel(
    const float*    __restrict__ featT,   // (BN,H,W,C)
    const float*    __restrict__ pn4,
    const float*    __restrict__ pxpy,
    const _Float16* __restrict__ wB,      // packed B fragments
    const float*    __restrict__ W1,      // (3,256)
    const float*    __restrict__ b1,      // (256)
    const float*    __restrict__ b2,      // (128)
    float*          __restrict__ feats_out)
{
    const int wave  = threadIdx.x >> 5;
    const int lane  = threadIdx.x & 31;
    const int group = blockIdx.x * 8 + wave;   // 3150 blocks * 8 waves = 25200
    const int n     = lane & 15;
    const int hi    = lane >> 4;
    const int pt0   = group * 16;

    // pn row for this lane's A-matrix row (M = n); lanes L and L+16 share it.
    const float4 p = *(const float4*)(pn4 + (size_t)(pt0 + n) * 4);

    v8f acc[8] = {};

    #pragma unroll
    for (int kc = 0; kc < 8; ++kc) {
        // ---- build fp16 A fragment: tiny layer-1 MLP in VALU ----
        v16h a;
        const int j0 = kc * 32 + hi * 8;       // half slots 0..7
        const int j1 = j0 + 16;                // half slots 8..15
        #pragma unroll
        for (int i = 0; i < 8; ++i) {
            float h0 = fmaf(p.x, W1[j0 + i],
                       fmaf(p.y, W1[256 + j0 + i],
                       fmaf(p.z, W1[512 + j0 + i], b1[j0 + i])));
            float h1 = fmaf(p.x, W1[j1 + i],
                       fmaf(p.y, W1[256 + j1 + i],
                       fmaf(p.z, W1[512 + j1 + i], b1[j1 + i])));
            a[i]     = (_Float16)fmaxf(h0, 0.f);
            a[i + 8] = (_Float16)fmaxf(h1, 0.f);
        }
        // ---- 8 N-tiles of WMMA against prepacked W2 fragments ----
        #pragma unroll
        for (int t = 0; t < 8; ++t) {
            const v16h bfrag =
                *(const v16h*)(wB + ((size_t)((kc * 8 + t) * 32 + lane) * 16));
            acc[t] = __builtin_amdgcn_wmma_f32_16x16x32_f16(
                false, a, false, bfrag, (short)0, acc[t], false, false);
        }
    }

    // ---- epilogue: + b2 + bilinear sample, coalesced store ----
    float b2v[8];
    #pragma unroll
    for (int t = 0; t < 8; ++t) b2v[t] = b2[t * 16 + n];

    #pragma unroll
    for (int r = 0; r < 8; ++r) {
        const int m     = hi * 8 + r;          // D: VGPR r holds M=r / M=r+8
        const int ptIdx = pt0 + m;
        const int bn    = ptIdx / (LL * P2);
        const float* fb = featT + (size_t)bn * (LL * CC);

        const float px = pxpy[(size_t)ptIdx * 2 + 0];
        const float py = pxpy[(size_t)ptIdx * 2 + 1];
        const float x0f = floorf(px), y0f = floorf(py);
        const float wx = px - x0f,   wy = py - y0f;
        const int xi0 = (int)x0f, yi0 = (int)y0f;
        const int xi1 = xi0 + 1,  yi1 = yi0 + 1;
        const float inx0 = (xi0 >= 0 && xi0 < WG) ? 1.f : 0.f;
        const float inx1 = (xi1 >= 0 && xi1 < WG) ? 1.f : 0.f;
        const float iny0 = (yi0 >= 0 && yi0 < HG) ? 1.f : 0.f;
        const float iny1 = (yi1 >= 0 && yi1 < HG) ? 1.f : 0.f;
        const int xc0 = iclamp(xi0, 0, WG - 1), xc1 = iclamp(xi1, 0, WG - 1);
        const int yc0 = iclamp(yi0, 0, HG - 1), yc1 = iclamp(yi1, 0, HG - 1);
        const float w00 = (1.f - wx) * (1.f - wy) * inx0 * iny0;
        const float w01 =        wx  * (1.f - wy) * inx1 * iny0;
        const float w10 = (1.f - wx) *        wy  * inx0 * iny1;
        const float w11 =        wx  *        wy  * inx1 * iny1;
        const float* p00 = fb + (size_t)(yc0 * WG + xc0) * CC;
        const float* p01 = fb + (size_t)(yc0 * WG + xc1) * CC;
        const float* p10 = fb + (size_t)(yc1 * WG + xc0) * CC;
        const float* p11 = fb + (size_t)(yc1 * WG + xc1) * CC;
        float* op = feats_out + (size_t)ptIdx * CC;

        #pragma unroll
        for (int t = 0; t < 8; ++t) {
            const int c = t * 16 + n;
            float smp = w00 * p00[c] + w01 * p01[c]
                      + w10 * p10[c] + w11 * p11[c];
            op[c] = acc[t][r] + b2v[t] + smp;
        }
    }
}

// ---------------------------------------------------------------------------
// Launcher
// ---------------------------------------------------------------------------
extern "C" void kernel_launch(void* const* d_in, const int* in_sizes, int n_in,
                              void* d_out, int out_size, void* d_ws, size_t ws_size,
                              hipStream_t stream) {
    const float* means3D = (const float*)d_in[0];
    const float* imgf    = (const float*)d_in[1];
    const float* dvec    = (const float*)d_in[2];
    const float* cov3D   = (const float*)d_in[3];
    const float* l2i     = (const float*)d_in[4];
    const float* W1      = (const float*)d_in[5];
    const float* b1      = (const float*)d_in[6];
    const float* W2      = (const float*)d_in[7];
    const float* b2      = (const float*)d_in[8];

    float* out   = (float*)d_out;
    float* feats = out + OUT_FEATS;
    float* cam   = out + OUT_CAM;
    float* pts   = out + OUT_PTS;

    char* ws = (char*)d_ws;
    float*     featT = (float*)(ws + WS_FEATT);
    float*     pn4   = (float*)(ws + WS_PN4);
    float*     pxpy  = (float*)(ws + WS_PXPY);
    _Float16*  wB    = (_Float16*)(ws + WS_WB);

    // 1) feature-map transpose: (BN,C,H,W) -> (BN,H,W,C)   [12*1680*128/256]
    transpose_feat_kernel<<<(BN * LL * CC) / 256, 256, 0, stream>>>(imgf, featT);
    // 2) pack W2 into fp16 WMMA B fragments                 [32768/256]
    pack_w2_kernel<<<128, 256, 0, stream>>>(W2, wB);
    // 3) geometry: pts / cam / pn4 / pixel coords
    geom_kernel<<<(NLOC + 255) / 256, 256, 0, stream>>>(
        means3D, dvec, cov3D, l2i, cam, pts, pn4, pxpy);
    // 4) WMMA MLP + bilinear sampling + fused add/store
    mlp_sample_kernel<<<NGROUPS / 8, 256, 0, stream>>>(
        featT, pn4, pxpy, wB, W1, b1, b2, feats);
}
